// InCritic_39994735461110
// MI455X (gfx1250) — matrix operand order repaired and checked by reading, not compiled, not run on hardware
//
#include <hip/hip_runtime.h>
#include <hip/hip_bf16.h>
#include <cstdint>

typedef _Float16 h16v __attribute__((ext_vector_type(16)));
typedef _Float16 h8v  __attribute__((ext_vector_type(8)));
typedef float    f8v  __attribute__((ext_vector_type(8)));
typedef unsigned int u32x4 __attribute__((ext_vector_type(4)));
typedef int          i32x4 __attribute__((ext_vector_type(4)));
typedef int          i32x8 __attribute__((ext_vector_type(8)));

#define WAVES 4
// ---- block LDS layout (bytes) ----
// weights blob (same order as workspace): 204800 B total
#define LW1_OFF 0         // 256x64 f16 = 32768
#define LW3_OFF 32768
#define LW2_OFF 65536     // 64x256 f16 = 32768
#define LW4_OFF 98304
#define LWV_OFF 131072    // 64x64 f16 = 8192
#define LR1_OFF 139264
#define LR3_OFF 172032
#define WEIGHT_BYTES 204800
#define WEIGHT_HALFS 102400
// per-wave scratch after weights
#define INP_OFF 0          // 16x64 f16  = 2048
#define HID_OFF 2048       // 16x256 f16 = 8192
#define X_OFF   10240      // 16x64 f16  = 2048
#define V_OFF   12288      // 5x16x64 f16= 10240
#define O_OFF   22528      // 16x64 f16  = 2048
#define QKW_OFF 24576      // q/k/w 3*16*5 f32 = 960
#define PER_WAVE 25600
#define LDS_TOTAL (WEIGHT_BYTES + WAVES * PER_WAVE)   // 307200 <= 320KB/WGP

struct Params {
    const float *obs, *act, *edge;
    const float *b1, *b2, *b3, *b4;
    const float *Wq, *bq, *Wk, *bk, *bv;
    const float *rb1, *R2, *rb2, *rb3, *R4, *rb4;
    const _Float16 *wsAll;   // contiguous f16 weight blob (WEIGHT_HALFS halfs)
    float* out;
    int B;
};

// A-operand (16x32 f16): lane<16 holds row M=lane, K = {k0+hi*16 .. +15}
__device__ __forceinline__ h16v ld_a_lds(const _Float16* buf, int strideH, int k0,
                                         int lrow, int lhi) {
    const int base = lrow * strideH + k0 + lhi * 16;
    return *(const h16v*)(buf + base);
}

// B-operand (32x16 f16) from f16 weights stored [N][K] in LDS: lane=N, 16 contiguous K halves
__device__ __forceinline__ h16v ld_b_lds(const _Float16* W, int strideH,
                                         int n0, int k0, int lrow, int lhi) {
    return *(const h16v*)(W + (n0 + lrow) * strideH + k0 + lhi * 16);
}

__device__ __forceinline__ f8v wmma_f16(h16v a, h16v b, f8v c) {
    return __builtin_amdgcn_wmma_f32_16x16x32_f16(false, a, false, b, (short)0, c,
                                                  false, false);
}

__device__ __forceinline__ f8v splat8(float v) {
    f8v c = {v, v, v, v, v, v, v, v};
    return c;
}

#if __has_builtin(__builtin_amdgcn_tensor_load_to_lds)
// TDM 2D load: rows x cols (f16 elements), contiguous, into LDS byte offset ldsOff.
// D# per CDNA5 ISA 8.3-8.6: group0 = {count, lds_addr, global_addr, type=2},
// group1 = {data_size=2B, tensor_dim0/1, tile_dim0/1, dim0_stride}; groups 2/3 zero (2D).
__device__ __forceinline__ void tdm_load_2d(const void* gsrc, unsigned ldsOff,
                                            unsigned cols, unsigned rows) {
    const unsigned long long ga = (unsigned long long)(uintptr_t)gsrc;
    u32x4 g0;
    g0.x = 1u;                                                 // count=1, user desc
    g0.y = ldsOff;                                             // lds_addr (bytes)
    g0.z = (unsigned)ga;                                       // global_addr[31:0]
    g0.w = (unsigned)((ga >> 32) & 0x1FFFFFFu) | (2u << 30);   // addr[56:32] | type=2
    i32x8 g1 = {0, 0, 0, 0, 0, 0, 0, 0};
    g1[0] = 0x00010000;                                        // data_size = 1 -> 2 bytes
    g1[1] = (int)((cols & 0xFFFFu) << 16);                     // tensor_dim0[15:0]
    g1[2] = (int)(((cols >> 16) & 0xFFFFu) | ((rows & 0xFFFFu) << 16)); // dim0 hi | dim1 lo
    g1[3] = (int)((cols & 0xFFFFu) << 16);                     // tile_dim0 (dim1 hi = 0)
    g1[4] = (int)(rows & 0xFFFFu);                             // tile_dim1
    g1[5] = (int)cols;                                         // tensor_dim0_stride[31:0]
    i32x4 gz = {0, 0, 0, 0};
#if __has_include(<hip/amd_detail/amd_gfx1250_TDM.h>)
    i32x8 gz8 = {0, 0, 0, 0, 0, 0, 0, 0};
    __builtin_amdgcn_tensor_load_to_lds(g0, g1, gz, gz, gz8, 0);
#else
    __builtin_amdgcn_tensor_load_to_lds(g0, g1, gz, gz, 0);
#endif
}
#endif

// weight convert: dst[n*Kpad+k] = (k<K) ? (f16)src[k*N+n] : 0
__global__ void cvt_w_kernel(const float* __restrict__ src, _Float16* __restrict__ dst,
                             int K, int N, int Kpad) {
    int idx = blockIdx.x * blockDim.x + threadIdx.x;
    if (idx >= N * Kpad) return;
    int n = idx / Kpad, k = idx % Kpad;
    dst[idx] = (k < K) ? (_Float16)src[(size_t)k * N + n] : (_Float16)0.f;
}

__global__ __launch_bounds__(WAVES * 32)
void InCritic_39994735461110_kernel(Params p) {
    extern __shared__ char smem[];
    const int wave = threadIdx.x >> 5;
    const int lane = threadIdx.x & 31;
    const int lrow = lane & 15;
    const int lhi  = lane >> 4;

    // ---- stage all f16 weights into LDS (once per block) ----
#if __has_builtin(__builtin_amdgcn_tensor_load_to_lds)
    if (wave == 0) {
        // whole 102400-half blob as 8 rows x 12800 halfs, one TDM op
        tdm_load_2d(p.wsAll, 0u, 12800u, 8u);
        __builtin_amdgcn_s_wait_tensorcnt(0);
    }
#else
    {
        const h8v* src = (const h8v*)p.wsAll;
        h8v* dst = (h8v*)smem;
        for (int i = threadIdx.x; i < WEIGHT_HALFS / 8; i += WAVES * 32)
            dst[i] = src[i];
    }
#endif
    __syncthreads();

    const _Float16* lW1 = (const _Float16*)(smem + LW1_OFF);
    const _Float16* lW3 = (const _Float16*)(smem + LW3_OFF);
    const _Float16* lW2 = (const _Float16*)(smem + LW2_OFF);
    const _Float16* lW4 = (const _Float16*)(smem + LW4_OFF);
    const _Float16* lWv = (const _Float16*)(smem + LWV_OFF);
    const _Float16* lR1 = (const _Float16*)(smem + LR1_OFF);
    const _Float16* lR3 = (const _Float16*)(smem + LR3_OFF);

    char* wb = smem + WEIGHT_BYTES + wave * PER_WAVE;
    _Float16* sInp = (_Float16*)(wb + INP_OFF);
    _Float16* sHid = (_Float16*)(wb + HID_OFF);
    _Float16* sX   = (_Float16*)(wb + X_OFF);
    _Float16* sV   = (_Float16*)(wb + V_OFF);
    _Float16* sO   = (_Float16*)(wb + O_OFF);
    float* sQ = (float*)(wb + QKW_OFF);  // [16][5]
    float* sK = sQ + 80;                 // [16][5]
    float* sW = sK + 80;                 // [16][5]

    const int B = p.B;
    const int rowBase = (blockIdx.x * WAVES + wave) * 16;

    for (int br = 0; br < 2; ++br) {
        const _Float16* Wp1 = br ? lW3 : lW1;
        const float*    bp1 = br ? p.b3 : p.b1;
        const _Float16* Wp2 = br ? lW4 : lW2;
        const float*    bp2 = br ? p.b4 : p.b2;
        const _Float16* Rt  = br ? lR3 : lR1;
        const float*    rbp = br ? p.rb3 : p.rb1;
        const float*    R2p = br ? p.R4  : p.R2;
        const float     rbq = br ? p.rb4[0] : p.rb2[0];

        for (int n = 0; n < 5; ++n) {
            // ---- build f16 input tile [16][64]: [act(4)|body(10)|obj(15)|edge(32)|pad]
            {
                const size_t rg = (size_t)(rowBase + lrow);
                for (int c = 0; c < 32; ++c) {
                    const int col = lhi * 32 + c;
                    float v;
                    if (col < 4)       v = p.act[rg * 4 + col];
                    else if (col < 14) v = p.obs[rg * 85 + (col - 4)];
                    else if (col < 29) v = p.obs[rg * 85 + 10 + n * 15 + (col - 14)];
                    else if (col < 61) v = p.edge[((size_t)n * B + rg) * 32 + (col - 29)];
                    else               v = 0.f;
                    sInp[lrow * 64 + col] = (_Float16)v;
                }
            }
            // ---- layer1: [16x64]@[64x256] + b, relu -> sHid
            h16v a0 = ld_a_lds(sInp, 64, 0,  lrow, lhi);
            h16v a1 = ld_a_lds(sInp, 64, 32, lrow, lhi);
            for (int nt = 0; nt < 16; ++nt) {
                const int n0 = nt * 16;
                f8v c = splat8(bp1[n0 + lrow]);
                c = wmma_f16(a0, ld_b_lds(Wp1, 64, n0, 0,  lrow, lhi), c);
                c = wmma_f16(a1, ld_b_lds(Wp1, 64, n0, 32, lrow, lhi), c);
#pragma unroll
                for (int r = 0; r < 8; ++r)
                    sHid[(lhi * 8 + r) * 256 + n0 + lrow] = (_Float16)fmaxf(c[r], 0.f);
            }
            // ---- layer2: [16x256]@[256x64] + b, relu -> sX
            for (int nt = 0; nt < 4; ++nt) {
                const int n0 = nt * 16;
                f8v c = splat8(bp2[n0 + lrow]);
#pragma unroll
                for (int kt = 0; kt < 8; ++kt) {
                    h16v a = ld_a_lds(sHid, 256, kt * 32, lrow, lhi);
                    c = wmma_f16(a, ld_b_lds(Wp2, 256, n0, kt * 32, lrow, lhi), c);
                }
#pragma unroll
                for (int r = 0; r < 8; ++r)
                    sX[(lhi * 8 + r) * 64 + n0 + lrow] = (_Float16)fmaxf(c[r], 0.f);
            }
            // ---- scalar q (lanes 0-15) / k (lanes 16-31) per row
            {
                const float* wqk = lhi ? p.Wk : p.Wq;
                float acc = lhi ? p.bk[0] : p.bq[0];
                for (int d = 0; d < 64; ++d)
                    acc += (float)sX[lrow * 64 + d] * wqk[d];
                if (lhi) sK[lrow * 5 + n] = acc;
                else     sQ[lrow * 5 + n] = acc;
            }
            // ---- v = x @ Wv + bv -> sV[n]  (no relu)
            _Float16* sVn = sV + n * 1024;
            h16v xa0 = ld_a_lds(sX, 64, 0,  lrow, lhi);
            h16v xa1 = ld_a_lds(sX, 64, 32, lrow, lhi);
            for (int nt = 0; nt < 4; ++nt) {
                const int n0 = nt * 16;
                f8v c = splat8(p.bv[n0 + lrow]);
                c = wmma_f16(xa0, ld_b_lds(lWv, 64, n0, 0,  lrow, lhi), c);
                c = wmma_f16(xa1, ld_b_lds(lWv, 64, n0, 32, lrow, lhi), c);
#pragma unroll
                for (int r = 0; r < 8; ++r)
                    sVn[(lhi * 8 + r) * 64 + n0 + lrow] = (_Float16)c[r];
            }
        } // objects

        // ---- softmax over 5x5 per row; w[m] = sum_nq attn[nq][m]
        if (lane < 16) {
            const int row = lane;
            float q[5], k[5], w[5] = {0.f, 0.f, 0.f, 0.f, 0.f};
#pragma unroll
            for (int i = 0; i < 5; ++i) { q[i] = sQ[row * 5 + i]; k[i] = sK[row * 5 + i]; }
#pragma unroll
            for (int nq = 0; nq < 5; ++nq) {
                float s[5], mx = -3.0e38f;
#pragma unroll
                for (int m = 0; m < 5; ++m) { s[m] = q[nq] * k[m]; mx = fmaxf(mx, s[m]); }
                float den = 0.f;
#pragma unroll
                for (int m = 0; m < 5; ++m) { s[m] = __expf(s[m] - mx); den += s[m]; }
                const float inv = 1.f / den;
#pragma unroll
                for (int m = 0; m < 5; ++m) w[m] += s[m] * inv;
            }
#pragma unroll
            for (int m = 0; m < 5; ++m) sW[row * 5 + m] = w[m];
        }
        // ---- o[row][col] = sum_m w[m] * v[m][row][col]  -> sO (f16)
        {
            float w[5];
#pragma unroll
            for (int m = 0; m < 5; ++m) w[m] = sW[lrow * 5 + m];
            for (int c = 0; c < 32; ++c) {
                const int col = lhi * 32 + c;
                float acc = 0.f;
#pragma unroll
                for (int m = 0; m < 5; ++m)
                    acc += w[m] * (float)sV[m * 1024 + lrow * 64 + col];
                sO[lrow * 64 + col] = (_Float16)acc;
            }
        }
        // ---- rho hidden: [16x64]@[64x256] + rb, relu -> sHid
        h16v oa0 = ld_a_lds(sO, 64, 0,  lrow, lhi);
        h16v oa1 = ld_a_lds(sO, 64, 32, lrow, lhi);
        for (int nt = 0; nt < 16; ++nt) {
            const int n0 = nt * 16;
            f8v c = splat8(rbp[n0 + lrow]);
            c = wmma_f16(oa0, ld_b_lds(Rt, 64, n0, 0,  lrow, lhi), c);
            c = wmma_f16(oa1, ld_b_lds(Rt, 64, n0, 32, lrow, lhi), c);
#pragma unroll
            for (int r = 0; r < 8; ++r)
                sHid[(lhi * 8 + r) * 256 + n0 + lrow] = (_Float16)fmaxf(c[r], 0.f);
        }
        // ---- final 256->1 dot per row
        if (lane < 16) {
            float acc = rbq;
            for (int j = 0; j < 256; ++j)
                acc += (float)sHid[lane * 256 + j] * R2p[j];
            p.out[(size_t)br * B + rowBase + lane] = acc;
        }
    } // branch
}

extern "C" void kernel_launch(void* const* d_in, const int* in_sizes, int n_in,
                              void* d_out, int out_size, void* d_ws, size_t ws_size,
                              hipStream_t stream) {
    (void)n_in; (void)out_size; (void)ws_size;
    const int B = in_sizes[1] / 4;  // act is (B,4)

    _Float16* ws  = (_Float16*)d_ws;
    _Float16* W1t = ws;              // 256*64
    _Float16* W3t = ws + 16384;      // 256*64
    _Float16* W2t = ws + 32768;      // 64*256
    _Float16* W4t = ws + 49152;      // 64*256
    _Float16* Wvt = ws + 65536;      // 64*64
    _Float16* R1t = ws + 69632;      // 256*64
    _Float16* R3t = ws + 86016;      // 256*64 -> blob ends at 102400 halfs

    auto cvt = [&](int src_idx, _Float16* dst, int K, int N, int Kpad) {
        int total = N * Kpad;
        cvt_w_kernel<<<(total + 255) / 256, 256, 0, stream>>>(
            (const float*)d_in[src_idx], dst, K, N, Kpad);
    };
    cvt(3,  W1t, 61,  256, 64);    // W1 (61,256)
    cvt(7,  W3t, 61,  256, 64);    // W3
    cvt(5,  W2t, 256, 64,  256);   // W2 (256,64)
    cvt(9,  W4t, 256, 64,  256);   // W4
    cvt(15, Wvt, 64,  64,  64);    // Wv (64,64)
    cvt(17, R1t, 64,  256, 64);    // R1 (64,256)
    cvt(21, R3t, 64,  256, 64);    // R3

    Params p;
    p.obs  = (const float*)d_in[0];
    p.act  = (const float*)d_in[1];
    p.edge = (const float*)d_in[2];
    p.b1   = (const float*)d_in[4];
    p.b2   = (const float*)d_in[6];
    p.b3   = (const float*)d_in[8];
    p.b4   = (const float*)d_in[10];
    p.Wq   = (const float*)d_in[11];
    p.bq   = (const float*)d_in[12];
    p.Wk   = (const float*)d_in[13];
    p.bk   = (const float*)d_in[14];
    p.bv   = (const float*)d_in[16];
    p.rb1  = (const float*)d_in[18];
    p.R2   = (const float*)d_in[19];
    p.rb2  = (const float*)d_in[20];
    p.rb3  = (const float*)d_in[22];
    p.R4   = (const float*)d_in[23];
    p.rb4  = (const float*)d_in[24];
    p.wsAll = ws;
    p.out = (float*)d_out;
    p.B = B;

    const int blocks = B / (WAVES * 16);
    InCritic_39994735461110_kernel<<<blocks, WAVES * 32, LDS_TOTAL, stream>>>(p);
}
